// Encoder_55327768707597
// MI455X (gfx1250) — compile-verified, matrix operand order checked by reading, not run on hardware
//
#include <hip/hip_runtime.h>
#include <hip/hip_bf16.h>
#include <math.h>

// Problem dims (fixed by reference)
#define BATCH 64
#define TSEQ  512
#define EDIM  512
#define HDIM  512
#define NG    (3 * HDIM)        // 1536
#define MROWS (BATCH * TSEQ)    // 32768

typedef __attribute__((ext_vector_type(16))) _Float16 v16h;
typedef __attribute__((ext_vector_type(8)))  _Float16 v8h;
typedef __attribute__((ext_vector_type(8)))  float    v8f;

// ---------------------------------------------------------------------------
// WMMA fragment helpers (wave32, V_WMMA_F32_16X16X32_F16)
//
// 16-bit A-matrix 16x32 layout (CDNA5 ISA §7.12.2):
//   lanes 0-15 : row M = lane,    VGPR0..3 = K 0..7,   VGPR4..7 = K 16..23
//   lanes 16-31: row M = lane-16, VGPR0..3 = K 8..15,  VGPR4..7 = K 24..31
// => per lane: two contiguous 8-half (16B) runs at K = kb and K = kb+16,
//    kb = kc + 8*(lane>=16).  B uses the mirrored layout with N = lane&15,
//    satisfied by pre-transposing B to [N][K] row-major.
// ---------------------------------------------------------------------------
__device__ __forceinline__ v16h load_frag16(const _Float16* __restrict__ P,
                                            int ld, int r0, int kc, int lane) {
  const int row = r0 + (lane & 15);
  const int kb  = kc + ((lane >> 4) << 3);
  const _Float16* p = P + (size_t)row * ld + kb;
  v8h lo = *reinterpret_cast<const v8h*>(p);        // global_load_b128
  v8h hi = *reinterpret_cast<const v8h*>(p + 16);   // global_load_b128
  return __builtin_shufflevector(lo, hi, 0, 1, 2, 3, 4, 5, 6, 7,
                                 8, 9, 10, 11, 12, 13, 14, 15);
}

__device__ __forceinline__ v8f wmma_f16(v16h a, v16h b, v8f c) {
  // 8 args: (neg_a, A, neg_b, B, c_mod, C, reuse_a, reuse_b)
  return __builtin_amdgcn_wmma_f32_16x16x32_f16(false, a, false, b,
                                                (short)0, c, false, false);
}

__device__ __forceinline__ float sigmoid_f(float x) {
  return 1.0f / (1.0f + __expf(-x));
}

// ---------------------------------------------------------------------------
// Prep kernels: f32 -> f16 convert and [K][N] -> [N][K] f16 transpose
// ---------------------------------------------------------------------------
__global__ void cvt_f32_to_f16(const float* __restrict__ src,
                               _Float16* __restrict__ dst, int n) {
  int i = blockIdx.x * blockDim.x + threadIdx.x;
  if (i < n) dst[i] = (_Float16)src[i];
}

__global__ void transpose_f32_to_f16(const float* __restrict__ W, // [K][N]
                                     _Float16* __restrict__ Wt,   // [N][K]
                                     int K, int N) {
  int i = blockIdx.x * blockDim.x + threadIdx.x;
  if (i < K * N) {
    int k = i / N;
    int ncol = i - k * N;
    Wt[(size_t)ncol * K + k] = (_Float16)W[i];
  }
}

__global__ void init_h(_Float16* __restrict__ h16, float* __restrict__ h32, int n) {
  int i = blockIdx.x * blockDim.x + threadIdx.x;
  if (i < n) { h16[i] = (_Float16)0.0f; h32[i] = 0.0f; }
}

// ---------------------------------------------------------------------------
// Phase 1: x_proj = src_seq @ Wx + b    (M=32768, K=512, N=1536)
// Register-blocked: each wave computes a 32x64 macro-tile (2 M-tiles x 4
// N-tiles -> 8 accumulators).  A fragments reused 4x, B fragments 2x:
// 12 b128-loads per 8 WMMAs.  All operand data is L2-resident (35 MB total
// vs 192 MB L2); the 201 MB x_proj result stream is written non-temporally
// so it does not evict the operand working set.
// ---------------------------------------------------------------------------
__global__ void __launch_bounds__(256)
xproj_gemm(const _Float16* __restrict__ A,   // [M][K]
           const _Float16* __restrict__ Bt,  // [N][K] (Wx transposed)
           const float* __restrict__ bias,   // [N]
           float* __restrict__ C) {          // [M][N]
  const int lane = threadIdx.x & 31;
  const int wave = blockIdx.x * (blockDim.x >> 5) + (threadIdx.x >> 5);
  const int NT4 = (NG / 16) / 4;       // 24 macro-columns
  const int tn4 = wave % NT4;
  const int tm2 = wave / NT4;          // 0..1023
  const int m0 = tm2 * 32;
  const int n0 = tn4 * 64;

  v8f acc[2][4] = {};
#pragma unroll 2
  for (int kc = 0; kc < EDIM; kc += 32) {
    v16h a0 = load_frag16(A, EDIM, m0,      kc, lane);
    v16h a1 = load_frag16(A, EDIM, m0 + 16, kc, lane);
#pragma unroll
    for (int j = 0; j < 4; ++j) {
      v16h b = load_frag16(Bt, EDIM, n0 + 16 * j, kc, lane);
      acc[0][j] = wmma_f16(a0, b, acc[0][j]);
      acc[1][j] = wmma_f16(a1, b, acc[1][j]);
    }
  }

  // C/D layout: VGPR i, lanes 0-15 -> (m0+i, col); lanes 16-31 -> (m0+8+i, col)
#pragma unroll
  for (int mi = 0; mi < 2; ++mi) {
    const int rbase = m0 + 16 * mi + ((lane >> 4) << 3);
#pragma unroll
    for (int j = 0; j < 4; ++j) {
      const int col  = n0 + 16 * j + (lane & 15);
      const float bv = bias[col];
#pragma unroll
      for (int i = 0; i < 8; ++i)
        __builtin_nontemporal_store(acc[mi][j][i] + bv,
                                    &C[(size_t)(rbase + i) * NG + col]);
    }
  }
}

// ---------------------------------------------------------------------------
// Phase 2: one GRU timestep (latency-bound scan step).
// rec = h_prev @ Wh ; gates fused in epilogue.  One wave per 16x16 tile of
// the H-wide output, with THREE accumulators over the z / r / h~ slices of
// Wh so the sigmoid/tanh blend is register-local.  Wh^T (1.5 MB f16) stays
// L2-resident across all 512 steps; x_proj is streamed in non-temporally.
// Kept un-blocked: step latency = one wave's 16-iteration WMMA chain.
// ---------------------------------------------------------------------------
__global__ void __launch_bounds__(256)
gru_step(const _Float16* __restrict__ hprev16, // [B][H] f16
         const float*    __restrict__ hprev32, // [B][H] f32
         const _Float16* __restrict__ WhT,     // [3H][H] (Wh transposed)
         const float*    __restrict__ xproj,   // [B*T][3H]
         int t,
         _Float16* __restrict__ hnext16,
         float*    __restrict__ hnext32,
         float*    __restrict__ hs) {          // [T][B][H]
  const int lane = threadIdx.x & 31;
  const int wave = blockIdx.x * (blockDim.x >> 5) + (threadIdx.x >> 5);
  const int jt = wave % (HDIM / 16);  // output-column tile within H
  const int mt = wave / (HDIM / 16);  // batch tile
  const int m0 = mt * 16, j0 = jt * 16;

  v8f az = {}, ar = {}, ah = {};
#pragma unroll 4
  for (int kc = 0; kc < HDIM; kc += 32) {
    v16h a  = load_frag16(hprev16, HDIM, m0, kc, lane);
    v16h bz = load_frag16(WhT, HDIM, j0,            kc, lane);
    v16h br = load_frag16(WhT, HDIM, HDIM + j0,     kc, lane);
    v16h bh = load_frag16(WhT, HDIM, 2 * HDIM + j0, kc, lane);
    az = wmma_f16(a, bz, az);
    ar = wmma_f16(a, br, ar);
    ah = wmma_f16(a, bh, ah);
  }

  const int col   = j0 + (lane & 15);
  const int rbase = m0 + ((lane >> 4) << 3);
#pragma unroll
  for (int i = 0; i < 8; ++i) {
    const int b = rbase + i;  // batch index
    const float* xp = xproj + (size_t)(b * TSEQ + t) * NG;
    const float xz = __builtin_nontemporal_load(&xp[col]);
    const float xr = __builtin_nontemporal_load(&xp[HDIM + col]);
    const float xh = __builtin_nontemporal_load(&xp[2 * HDIM + col]);
    const float z  = sigmoid_f(xz + az[i]);
    const float r  = sigmoid_f(xr + ar[i]);
    const float hh = tanhf(xh + r * ah[i]);
    const float hp = hprev32[(size_t)b * HDIM + col];
    const float hn = z * hp + (1.0f - z) * hh;
    hnext32[(size_t)b * HDIM + col] = hn;
    hnext16[(size_t)b * HDIM + col] = (_Float16)hn;
    __builtin_nontemporal_store(hn,
        &hs[(size_t)t * BATCH * HDIM + (size_t)b * HDIM + col]);
  }
}

__global__ void write_last(const float* __restrict__ hfin, float* __restrict__ out) {
  int i = blockIdx.x * blockDim.x + threadIdx.x;
  if (i < BATCH * HDIM) {
    out[i] = hfin[i];                                                      // outputs[0]
    out[(size_t)BATCH * HDIM + (size_t)TSEQ * BATCH * HDIM + i] = hfin[i]; // outputs[2]
  }
}

// ---------------------------------------------------------------------------
// Host-side launch
// ---------------------------------------------------------------------------
extern "C" void kernel_launch(void* const* d_in, const int* in_sizes, int n_in,
                              void* d_out, int out_size, void* d_ws, size_t ws_size,
                              hipStream_t stream) {
  (void)in_sizes; (void)n_in; (void)out_size; (void)ws_size;

  const float* src  = (const float*)d_in[0]; // [B][T][E]
  const float* Wx   = (const float*)d_in[1]; // [E][3H]
  const float* Wh   = (const float*)d_in[2]; // [H][3H]
  const float* bias = (const float*)d_in[3]; // [3H]
  float* out = (float*)d_out;                // last | hs[T][B][H] | last

  // Workspace layout
  char* p = (char*)d_ws;
  _Float16* srcF16 = (_Float16*)p; p += (size_t)MROWS * EDIM * sizeof(_Float16); // 33.5 MB
  _Float16* WxT    = (_Float16*)p; p += (size_t)NG * EDIM * sizeof(_Float16);    // 1.5 MB
  _Float16* WhT    = (_Float16*)p; p += (size_t)NG * HDIM * sizeof(_Float16);    // 1.5 MB
  float*    xproj  = (float*)p;    p += (size_t)MROWS * NG * sizeof(float);      // 201 MB
  _Float16* hF16   = (_Float16*)p; p += (size_t)2 * BATCH * HDIM * sizeof(_Float16);
  float*    hF32   = (float*)p;    p += (size_t)2 * BATCH * HDIM * sizeof(float);

  // Prep: convert/transpose into f16
  { int n = MROWS * EDIM;
    cvt_f32_to_f16<<<(n + 255) / 256, 256, 0, stream>>>(src, srcF16, n); }
  { int n = EDIM * NG;
    transpose_f32_to_f16<<<(n + 255) / 256, 256, 0, stream>>>(Wx, WxT, EDIM, NG); }
  { int n = HDIM * NG;
    transpose_f32_to_f16<<<(n + 255) / 256, 256, 0, stream>>>(Wh, WhT, HDIM, NG); }

  // Phase 1: big GEMM.  32x64 macro-tile per wave; 8 waves / 256-thread block.
  { int waves = (MROWS / 32) * (NG / 64);     // 1024 * 24 = 24576
    xproj_gemm<<<waves / 8, 256, 0, stream>>>(srcF16, WxT, bias, xproj); }

  // h0 = 0
  { int n = BATCH * HDIM;
    init_h<<<(n + 255) / 256, 256, 0, stream>>>(hF16, hF32, n); }

  // Phase 2: sequential scan, ping-pong h buffers, one launch per step
  float* hs = out + (size_t)BATCH * HDIM;
  for (int t = 0; t < TSEQ; ++t) {
    const _Float16* hp16 = hF16 + (size_t)(t & 1) * BATCH * HDIM;
    const float*    hp32 = hF32 + (size_t)(t & 1) * BATCH * HDIM;
    _Float16* hn16 = hF16 + (size_t)((t + 1) & 1) * BATCH * HDIM;
    float*    hn32 = hF32 + (size_t)((t + 1) & 1) * BATCH * HDIM;
    // (B/16)*(H/16) = 128 waves -> 16 blocks of 256 threads
    gru_step<<<16, 256, 0, stream>>>(hp16, hp32, WhT, xproj, t, hn16, hn32, hs);
  }

  // Final hidden state lives in buffer (TSEQ & 1) == 0
  { int n = BATCH * HDIM;
    write_last<<<(n + 255) / 256, 256, 0, stream>>>(hF32, out); }
}
